// SFDIVEQ_53017076302228
// MI455X (gfx1250) — matrix operand order repaired and checked by reading, not compiled, not run on hardware
//
#include <hip/hip_runtime.h>

typedef __attribute__((ext_vector_type(16))) __bf16 v16bf;
typedef __attribute__((ext_vector_type(8)))  float  v8f;
typedef __attribute__((ext_vector_type(4)))  unsigned int u32x4;
typedef __attribute__((ext_vector_type(8)))  int i32x8;
typedef __attribute__((ext_vector_type(4)))  int i32x4;

#define D64   64
#define KCB   1024          // padded column count (1023 real dithered codes)
#define NTILE 64            // KCB / 16
#define BPACK_BYTES 131072  // 64 tiles * 32 lanes * 2 frags * 16 bf16 * 2B
#define STAGE_BYTES 135168  // BPACK_BYTES + 1024 offsets * 4B (contiguous)

// ---------------------------------------------------------------------------
// Kernel 1: build dithered codebook in WMMA B-fragment bf16 layout,
// per-column score offsets (-0.5*||c||^2), and zero the histogram counts.
// B packing used here (contiguous k, same as A): lane l -> column n = l&15, hi = l>>4,
// element e (0..15) -> K = w*32 + hi*16 + e  (w = which of the two 32-K blocks)
// ---------------------------------------------------------------------------
__global__ __launch_bounds__(256) void vq_prep(
    const float* __restrict__ codebook,   // [1024][64]
    const float* __restrict__ dither,     // [1023]
    __bf16* __restrict__ bpack,           // [64 tiles][32 lanes][2 frags][16]
    float*  __restrict__ offs,            // [1024]
    float*  __restrict__ counts)          // [1024]
{
  const int t   = blockIdx.x;     // tile 0..63
  const int tid = threadIdx.x;    // 0..255
  #pragma unroll
  for (int q = 0; q < 4; ++q) {
    int f  = tid * 4 + q;         // flat 0..1023 within tile
    int e  = f & 15;
    int w  = (f >> 4) & 1;
    int l  = f >> 5;              // lane 0..31
    int n  = t * 16 + (l & 15);   // column index
    int hi = l >> 4;
    int k  = w * 32 + hi * 16 + e;
    float val = 0.0f;
    if (n < KCB - 1) {
      float lo = codebook[n * D64 + k];
      float hc = codebook[(n + 1) * D64 + k];
      val = lo + dither[n] * (hc - lo);
    }
    bpack[((t * 32 + l) * 2 + w) * 16 + e] = (__bf16)val;
  }
  if (tid < 16) {
    int n = t * 16 + tid;
    float o;
    if (n < KCB - 1) {
      float s = 0.0f, dv = dither[n];
      for (int k = 0; k < D64; ++k) {
        float lo = codebook[n * D64 + k];
        float hc = codebook[(n + 1) * D64 + k];
        float v  = lo + dv * (hc - lo);
        s += v * v;
      }
      o = -0.5f * s;
    } else {
      o = -1.0e30f;               // padded column can never win the argmax
    }
    offs[n] = o;
  }
  if (t == 0) {
    for (int i = tid; i < KCB; i += 256) counts[i] = 0.0f;
  }
}

// ---------------------------------------------------------------------------
// Kernel 2: score GEMM via v_wmma_f32_16x16x32_bf16 + running argmax.
// WG = 256 threads = 8 waves; each wave owns TWO 16-row tiles -> 256 rows/WG.
// Packed B table + offsets (132KB) staged into LDS by the Tensor Data Mover
// (single 1-D descriptor, data_size=8B), with cooperative-copy fallback.
// ---------------------------------------------------------------------------
__global__ __launch_bounds__(256, 1) void vq_argmax(
    const float* __restrict__ z,          // [N][64]
    const void*  __restrict__ bstage,     // bpack(131072B) + offs(4096B)
    int*         __restrict__ out_idx)    // [N]
{
  __shared__ __align__(16) unsigned char smem[STAGE_BYTES];
  const v16bf* sBf  = (const v16bf*)smem;             // B fragments (32B each)
  const float* sOff = (const float*)(smem + BPACK_BYTES);

  const int tid  = threadIdx.x;
  const int wave = tid >> 5;
  const int lane = tid & 31;
  const int hi   = lane >> 4;
  const int lm   = lane & 15;

#if __has_builtin(__builtin_amdgcn_tensor_load_to_lds) && \
    __has_builtin(__builtin_amdgcn_s_wait_tensorcnt)
  if (wave == 0) {
    const unsigned long long ga = (unsigned long long)bstage;
    const unsigned lds  = (unsigned)(uintptr_t)&smem[0]; // low 32b = LDS offset
    const unsigned nel  = STAGE_BYTES / 8;               // 16896 x 8B elements
    // D# group 0: count=1 | lds_addr | 57-bit global_addr | type=2
    u32x4 g0;
    g0[0] = 1u;
    g0[1] = lds;
    g0[2] = (unsigned)(ga & 0xFFFFFFFFull);
    g0[3] = ((unsigned)(ga >> 32) & 0x01FFFFFFu) | (2u << 30);
    // D# group 1: data_size=8B; tensor_dim0=tile_dim0=16896; dims1..=1/unused
    i32x8 g1;
    g1[0] = (int)(3u << 16);                 // data_size = 3 (8 bytes)
    g1[1] = (int)((nel & 0xFFFFu) << 16);    // tensor_dim0[15:0]  @ bits 63:48
    g1[2] = (int)((nel >> 16) | (1u << 16)); // tensor_dim0[31:16]; tensor_dim1=1
    g1[3] = (int)((nel & 0xFFFFu) << 16);    // tile_dim0 @ bits 127:112
    g1[4] = 1;                               // tile_dim1 = 1
    g1[5] = (int)nel;                        // tensor_dim0_stride[31:0]
    g1[6] = 0;
    g1[7] = 0;
    i32x4 gz4 = {0, 0, 0, 0};
    i32x8 gz8 = {0, 0, 0, 0, 0, 0, 0, 0};
    __builtin_amdgcn_tensor_load_to_lds(g0, g1, gz4, gz4, gz8, 0);
    __builtin_amdgcn_s_wait_tensorcnt(0);
  }
  __syncthreads();
#else
  {
    uint4* sV = (uint4*)smem;
    const uint4* gV = (const uint4*)bstage;
    for (int i = tid; i < STAGE_BYTES / 16; i += 256) sV[i] = gV[i];
    __syncthreads();
  }
#endif

  // --- load A fragments (z rows, bf16) once; reused across all 64 col tiles.
  // A order used here (contiguous k, same as B packing): lane holds row M=lm; e0..15 -> K = hi*16+e;
  // second fragment adds K+=32.
  v16bf a[2][2];                       // [row-tile][k-block]
  #pragma unroll
  for (int rt = 0; rt < 2; ++rt) {
    const long arow = (long)blockIdx.x * 256 + wave * 32 + rt * 16 + lm;
    const float* zr = z + arow * D64;
    #pragma unroll
    for (int w = 0; w < 2; ++w) {
      const int base = w * 32 + hi * 16;
      float4 p0 = *(const float4*)(zr + base);
      float4 p1 = *(const float4*)(zr + base + 4);
      float4 p2 = *(const float4*)(zr + base + 8);
      float4 p3 = *(const float4*)(zr + base + 12);
      v16bf v;
      v[0]=(__bf16)p0.x; v[1]=(__bf16)p0.y; v[2]=(__bf16)p0.z; v[3]=(__bf16)p0.w;
      v[4]=(__bf16)p1.x; v[5]=(__bf16)p1.y; v[6]=(__bf16)p1.z; v[7]=(__bf16)p1.w;
      v[8]=(__bf16)p2.x; v[9]=(__bf16)p2.y; v[10]=(__bf16)p2.z; v[11]=(__bf16)p2.w;
      v[12]=(__bf16)p3.x; v[13]=(__bf16)p3.y; v[14]=(__bf16)p3.z; v[15]=(__bf16)p3.w;
      a[rt][w] = v;
    }
  }

  float best[2][8];
  int   bidx[2][8];
  #pragma unroll
  for (int rt = 0; rt < 2; ++rt)
    #pragma unroll
    for (int v = 0; v < 8; ++v) { best[rt][v] = -3.0e38f; bidx[rt][v] = 0; }

  for (int t = 0; t < NTILE; ++t) {
    v16bf b0 = sBf[(t * 32 + lane) * 2 + 0];
    v16bf b1 = sBf[(t * 32 + lane) * 2 + 1];
    const int   n   = t * 16 + lm;      // this lane's column (C/D: N = lane%16)
    const float off = sOff[n];
    #pragma unroll
    for (int rt = 0; rt < 2; ++rt) {
      v8f acc = {0.f,0.f,0.f,0.f,0.f,0.f,0.f,0.f};
      acc = __builtin_amdgcn_wmma_f32_16x16x32_bf16(false, a[rt][0], false, b0,
                                                    (short)0, acc, false, false);
      acc = __builtin_amdgcn_wmma_f32_16x16x32_bf16(false, a[rt][1], false, b1,
                                                    (short)0, acc, false, false);
      #pragma unroll
      for (int v = 0; v < 8; ++v) {
        float s = acc[v] + off;
        if (s > best[rt][v]) { best[rt][v] = s; bidx[rt][v] = n; }
      }
    }
  }

  // reduce argmax across the 16 lanes sharing each row (halves are independent:
  // lanes 0-15 hold rows 0-7, lanes 16-31 hold rows 8-15)
  #pragma unroll
  for (int m = 1; m < 16; m <<= 1) {
    #pragma unroll
    for (int rt = 0; rt < 2; ++rt) {
      #pragma unroll
      for (int v = 0; v < 8; ++v) {
        float os = __shfl_xor(best[rt][v], m, 16);
        int   oi = __shfl_xor(bidx[rt][v], m, 16);
        if (os > best[rt][v] || (os == best[rt][v] && oi < bidx[rt][v])) {
          best[rt][v] = os; bidx[rt][v] = oi;
        }
      }
    }
  }
  if (lm == 0) {
    #pragma unroll
    for (int rt = 0; rt < 2; ++rt) {
      const long rbase = (long)blockIdx.x * 256 + wave * 32 + rt * 16 + 8 * hi;
      #pragma unroll
      for (int v = 0; v < 8; ++v) out_idx[rbase + v] = bidx[rt][v];
    }
  }
}

// ---------------------------------------------------------------------------
// Kernel 3: element-wise post pass (exact f32). One wave per row; lane owns
// 2 elements; row norms via wave32 shuffle reduction.
// ---------------------------------------------------------------------------
__global__ __launch_bounds__(256) void vq_post(
    const float* __restrict__ z,
    const float* __restrict__ codebook,
    const float* __restrict__ dither,
    const float* __restrict__ noise1,
    const float* __restrict__ noise2,
    const int*   __restrict__ indices,
    float* __restrict__ zq_out,
    float* __restrict__ idx_out,
    float* __restrict__ counts)
{
  const int wave = threadIdx.x >> 5;
  const int lane = threadIdx.x & 31;
  const long row = (long)blockIdx.x * 8 + wave;
  const int  idx = indices[row];
  const float lam = dither[idx];

  const long zo = row * D64 + lane * 2;
  float2 zv  = *(const float2*)(z + zo);
  float2 cb1 = *(const float2*)(codebook + (long)idx * D64 + lane * 2);
  float2 cb2 = *(const float2*)(codebook + (long)(idx + 1) * D64 + lane * 2);
  float2 n1  = *(const float2*)(noise1 + zo);
  float2 n2  = *(const float2*)(noise2 + zo);

  float2 d1  = { cb1.x - zv.x, cb1.y - zv.y };
  float2 d2  = { cb2.x - zv.x, cb2.y - zv.y };
  float2 rv1 = { n1.x + d1.x,  n1.y + d1.y };
  float2 rv2 = { n2.x + d2.x,  n2.y + d2.y };

  float sd1 = d1.x * d1.x + d1.y * d1.y;
  float sd2 = d2.x * d2.x + d2.y * d2.y;
  float sr1 = rv1.x * rv1.x + rv1.y * rv1.y;
  float sr2 = rv2.x * rv2.x + rv2.y * rv2.y;
  #pragma unroll
  for (int m = 1; m < 32; m <<= 1) {
    sd1 += __shfl_xor(sd1, m, 32);
    sd2 += __shfl_xor(sd2, m, 32);
    sr1 += __shfl_xor(sr1, m, 32);
    sr2 += __shfl_xor(sr2, m, 32);
  }
  const float w1 = sqrtf(sd1) * (1.0f - lam) / fmaxf(sqrtf(sr1), 1e-12f);
  const float w2 = sqrtf(sd2) * lam          / fmaxf(sqrtf(sr2), 1e-12f);
  float2 out = { zv.x + w1 * rv1.x + w2 * rv2.x,
                 zv.y + w1 * rv1.y + w2 * rv2.y };
  *(float2*)(zq_out + zo) = out;

  if (lane == 0) {
    idx_out[row] = (float)idx;
    atomicAdd(&counts[idx], 1.0f);
  }
}

// ---------------------------------------------------------------------------
// Kernel 4: perplexity = exp(-sum p log p) over the 1024-bin histogram.
// ---------------------------------------------------------------------------
__global__ __launch_bounds__(256) void vq_perplexity(
    const float* __restrict__ counts, float* __restrict__ out, float inv_n)
{
  __shared__ float red[8];
  const int tid = threadIdx.x;
  float h = 0.0f;
  for (int i = tid; i < KCB; i += 256) {
    float p = counts[i] * inv_n;
    if (p > 0.0f) h += p * logf(p);
  }
  #pragma unroll
  for (int m = 1; m < 32; m <<= 1) h += __shfl_xor(h, m, 32);
  if ((tid & 31) == 0) red[tid >> 5] = h;
  __syncthreads();
  if (tid == 0) {
    float s = 0.0f;
    #pragma unroll
    for (int i = 0; i < 8; ++i) s += red[i];
    out[0] = expf(-s);
  }
}

// ---------------------------------------------------------------------------
extern "C" void kernel_launch(void* const* d_in, const int* in_sizes, int n_in,
                              void* d_out, int out_size, void* d_ws, size_t ws_size,
                              hipStream_t stream) {
  const float* z        = (const float*)d_in[0];
  const float* codebook = (const float*)d_in[1];
  const float* dither   = (const float*)d_in[2];
  const float* noise1   = (const float*)d_in[3];
  const float* noise2   = (const float*)d_in[4];

  const int N = in_sizes[0] / D64;        // 131072

  float* zq      = (float*)d_out;
  float* idx_out = zq + (size_t)N * D64;
  float* ppl     = idx_out + N;

  char*   ws      = (char*)d_ws;
  __bf16* bpack   = (__bf16*)(ws);                 // 131072 B
  float*  offs    = (float*)(ws + BPACK_BYTES);    //   4096 B (contiguous)
  float*  counts  = (float*)(ws + STAGE_BYTES);    //   4096 B
  int*    indices = (int*)(ws + STAGE_BYTES + 4096); // N*4 B

  vq_prep<<<NTILE, 256, 0, stream>>>(codebook, dither, bpack, offs, counts);
  vq_argmax<<<N / 256, 256, 0, stream>>>(z, (const void*)ws, indices);
  vq_post<<<N / 8, 256, 0, stream>>>(z, codebook, dither, noise1, noise2,
                                     indices, zq, idx_out, counts);
  vq_perplexity<<<1, 256, 0, stream>>>(counts, ppl, 1.0f / (float)N);
}
